// HGT_541165879645
// MI455X (gfx1250) — compile-verified, matrix operand order checked by reading, not run on hardware
//
#include <hip/hip_runtime.h>
#include <math.h>

// ---------------------------------------------------------------------------
// HGT forward for MI455X (gfx1250, wave32).
// Dense math -> v_wmma_f32_16x16x32_bf16 (bf16 in, f32 accum), wave owns a
//               16x128 output strip: A fragment loaded/converted once per
//               K-step, 8 independent WMMAs against packed B fragments.
// Edge math  -> wave-per-edge coalesced gathers + HW f32 atomics.
// Params assumed flattened in setup_inputs() dict insertion order.
// ---------------------------------------------------------------------------

#define DEVI __device__ __forceinline__

typedef __attribute__((ext_vector_type(16))) __bf16 v16bf;
typedef __attribute__((ext_vector_type(8)))  float  v8f;

union ABFrag {
  v16bf v;
  uint4 u[2];
  unsigned short s[16];
};

// native RNE f32 -> bf16 (let clang pick the hw conversion op)
DEVI unsigned short f2bf_bits(float f) {
  union { __bf16 b; unsigned short s; } u;
  u.b = (__bf16)f;
  return u.s;
}
// order-preserving float -> uint encoding (for atomicMax-based segment max)
DEVI unsigned encF(float f) {
  unsigned u = __float_as_uint(f);
  return (u & 0x80000000u) ? ~u : (u | 0x80000000u);
}
DEVI float decF(unsigned e) {
  unsigned u = (e & 0x80000000u) ? (e & 0x7FFFFFFFu) : ~e;
  return __uint_as_float(u);
}
DEVI float gelu_exact(float x) {
  return 0.5f * x * (1.0f + erff(x * 0.70710678118654752440f));
}

// ---------------------------------------------------------------------------
// Pack a row-major f32 weight [K,N] into bf16 WMMA B-fragment order.
// Fragment layout (per (kb,ntile) tile): lane holds col = t*16+(lane&15),
// K elements e<8 -> kb*32 + (lane>>4)*8 + e ; e>=8 -> +16.
// grid = (K/32)*(N/16) blocks of 32 threads.
// ---------------------------------------------------------------------------
__global__ void pack_w_bf16(const float* __restrict__ W,
                            unsigned short* __restrict__ out, int N) {
  const int lane = threadIdx.x;
  const int ntiles = N >> 4;
  const int kb = blockIdx.x / ntiles;
  const int t  = blockIdx.x % ntiles;
  const int hi = lane >> 4;
  const int col = t * 16 + (lane & 15);
  unsigned short* o = out + ((size_t)blockIdx.x * 32 + lane) * 16;
#pragma unroll
  for (int e = 0; e < 16; ++e) {
    int kk = kb * 32 + hi * 8 + (e < 8 ? e : e + 8);
    o[e] = f2bf_bits(W[(size_t)kk * N + col]);
  }
}

// ---------------------------------------------------------------------------
// WMMA GEMM: out[M,128] = epilogue(A[M,K] @ B[K,128] + bias).
// One wave per 16-row strip: 8 f32 accumulators (full 128-wide output),
// A fragment converted to bf16 once per K-step, 8 WMMAs per K-step.
// 8 waves per block -> 8 strips per block; tail strips guarded.
// epi: 0 = bias, 1 = bias+relu, 2 = bias + sigmoid(skip)-mix with hOld.
// ---------------------------------------------------------------------------
__global__ __launch_bounds__(256) void hgt_gemm(
    const float* __restrict__ A, int K, int M,
    const unsigned short* __restrict__ Bp,
    const float* __restrict__ bias,
    float* __restrict__ out,
    const float* __restrict__ hOld,
    const float* __restrict__ skipP,
    int epi) {
  const int lane  = threadIdx.x & 31;
  const int wave  = threadIdx.x >> 5;
  const int mtile = blockIdx.x * 8 + wave;
  if (mtile * 16 >= M) return;
  const int hi  = lane >> 4;
  const int row = mtile * 16 + (lane & 15);
  const float* Arow = A + (size_t)row * K + hi * 8;
  const uint4* Bq = (const uint4*)Bp;
  v8f acc[8];
#pragma unroll
  for (int nt = 0; nt < 8; ++nt) acc[nt] = (v8f){};
  const int nkb = K >> 5;
  for (int kb = 0; kb < nkb; ++kb) {
    ABFrag a;
    const float* ap = Arow + kb * 32;
#pragma unroll
    for (int e = 0; e < 16; ++e) a.v[e] = (__bf16)ap[e < 8 ? e : e + 8];
#pragma unroll
    for (int nt = 0; nt < 8; ++nt) {
      ABFrag b;
      const size_t idx = (((size_t)(kb * 8 + nt)) * 32 + lane) * 2;
      b.u[0] = Bq[idx];
      b.u[1] = Bq[idx + 1];
      acc[nt] = __builtin_amdgcn_wmma_f32_16x16x32_bf16(
          false, a.v, false, b.v, (short)0, acc[nt], false, false);
    }
  }
  float sk = 0.f;
  if (epi == 2) sk = 1.f / (1.f + __expf(-skipP[0]));
#pragma unroll
  for (int nt = 0; nt < 8; ++nt) {
    const int col = nt * 16 + (lane & 15);
    const float bv = bias[col];
#pragma unroll
    for (int r = 0; r < 8; ++r) {
      int rr = mtile * 16 + hi * 8 + r;
      float vv = acc[nt][r] + bv;
      if (epi == 1) vv = fmaxf(vv, 0.f);
      if (epi == 2) vv = sk * vv + (1.f - sk) * hOld[(size_t)rr * 128 + col];
      out[(size_t)rr * 128 + col] = vv;
    }
  }
}

// ---------------------------------------------------------------------------
// Per-head relation transform: Krel[n, h*32+e] = sum_d K[n, h*32+d]*arel[h,d,e]
// (and Vrel with mrel). 8 waves/block: wave = head*2 + n-half-tile.
// Single 16x16x32 WMMA per matrix per wave.
// ---------------------------------------------------------------------------
__global__ __launch_bounds__(256) void hgt_rel(
    const float* __restrict__ Kf, const float* __restrict__ Vf,
    const unsigned short* __restrict__ ArelP,
    const unsigned short* __restrict__ MrelP,
    float* __restrict__ Krel, float* __restrict__ Vrel) {
  const int lane = threadIdx.x & 31;
  const int wave = threadIdx.x >> 5;
  const int head = wave >> 1;
  const int t    = wave & 1;
  const int hi   = lane >> 4;
  const int row  = blockIdx.x * 16 + (lane & 15);
  const size_t abase = (size_t)row * 128 + head * 32 + hi * 8;
  ABFrag ka, va, kb, vb;
#pragma unroll
  for (int e = 0; e < 16; ++e) {
    int o = (e < 8 ? e : e + 8);
    ka.v[e] = (__bf16)Kf[abase + o];
    va.v[e] = (__bf16)Vf[abase + o];
  }
  const uint4* bq =
      (const uint4*)(ArelP + (size_t)head * 1024 + ((size_t)(t * 32 + lane)) * 16);
  kb.u[0] = bq[0];
  kb.u[1] = bq[1];
  const uint4* mq =
      (const uint4*)(MrelP + (size_t)head * 1024 + ((size_t)(t * 32 + lane)) * 16);
  vb.u[0] = mq[0];
  vb.u[1] = mq[1];
  v8f acck = {}, accv = {};
  acck = __builtin_amdgcn_wmma_f32_16x16x32_bf16(false, ka.v, false, kb.v,
                                                 (short)0, acck, false, false);
  accv = __builtin_amdgcn_wmma_f32_16x16x32_bf16(false, va.v, false, vb.v,
                                                 (short)0, accv, false, false);
  const int col = head * 32 + t * 16 + (lane & 15);
#pragma unroll
  for (int r = 0; r < 8; ++r) {
    int rr = blockIdx.x * 16 + hi * 8 + r;
    Krel[(size_t)rr * 128 + col] = acck[r];
    Vrel[(size_t)rr * 128 + col] = accv[r];
  }
}

// ---------------------------------------------------------------------------
// Edge pass A: alpha[e,h] = SCALE * prel[h] * <q[dst],krel[src]>_head,
// atomic segment-max into mEnc[dst,h]. One wave per edge, float4 per lane.
// ---------------------------------------------------------------------------
__global__ __launch_bounds__(256) void hgt_edge_alpha(
    const int* __restrict__ src, const int* __restrict__ dst,
    const float* __restrict__ Q, const float* __restrict__ Krel,
    const float* __restrict__ prel, float* __restrict__ alpha,
    unsigned* __restrict__ mEnc, int E) {
  int e = blockIdx.x * 8 + (threadIdx.x >> 5);
  if (e >= E) return;
  const int lane = threadIdx.x & 31;
  const int s = src[e], d = dst[e];
  float4 qv = ((const float4*)(Q + (size_t)d * 128))[lane];
  float4 kv = ((const float4*)(Krel + (size_t)s * 128))[lane];
  float p = qv.x * kv.x + qv.y * kv.y + qv.z * kv.z + qv.w * kv.w;
  p += __shfl_xor(p, 1);
  p += __shfl_xor(p, 2);
  p += __shfl_xor(p, 4);
  if ((lane & 7) == 0) {
    int h = lane >> 3;
    float a = p * prel[h] * 0.17677669529663687f;  // 1/sqrt(32)
    alpha[(size_t)e * 4 + h] = a;
    atomicMax(&mEnc[(size_t)d * 4 + h], encF(a));
  }
}

// Edge pass B: ex = exp(alpha - m[dst]); atomic sum into sbuf[dst,h].
__global__ void hgt_edge_expsum(const int* __restrict__ dst,
                                float* __restrict__ alpha,
                                const unsigned* __restrict__ mEnc,
                                float* __restrict__ sbuf, int E) {
  int t = blockIdx.x * blockDim.x + threadIdx.x;
  if (t >= E * 4) return;
  int e = t >> 2, h = t & 3;
  int d = dst[e];
  float m = decF(mEnc[(size_t)d * 4 + h]);
  float ex = __expf(alpha[t] - m);
  alpha[t] = ex;
  unsafeAtomicAdd(&sbuf[(size_t)d * 4 + h], ex);
}

// Edge pass C: agg[dst] += vrel[src] * ex/(sum+1e-16). One wave per edge.
__global__ __launch_bounds__(256) void hgt_edge_scatter(
    const int* __restrict__ src, const int* __restrict__ dst,
    const float* __restrict__ Vrel, const float* __restrict__ alpha,
    const float* __restrict__ sbuf, float* __restrict__ agg, int E) {
  int e = blockIdx.x * 8 + (threadIdx.x >> 5);
  if (e >= E) return;
  const int lane = threadIdx.x & 31;
  const int s = src[e], d = dst[e];
  const int h = lane >> 3;
  float coef = alpha[(size_t)e * 4 + h] / (sbuf[(size_t)d * 4 + h] + 1e-16f);
  float4 vv = ((const float4*)(Vrel + (size_t)s * 128))[lane];
  float* ap = agg + (size_t)d * 128 + lane * 4;
  unsafeAtomicAdd(ap + 0, vv.x * coef);
  unsafeAtomicAdd(ap + 1, vv.y * coef);
  unsafeAtomicAdd(ap + 2, vv.z * coef);
  unsafeAtomicAdd(ap + 3, vv.w * coef);
}

__global__ void gelu_inplace(float* __restrict__ p, int n) {
  int i = blockIdx.x * blockDim.x + threadIdx.x;
  if (i < n) p[i] = gelu_exact(p[i]);
}

__global__ void fill_u32(unsigned* __restrict__ p, unsigned v, int n) {
  int i = blockIdx.x * blockDim.x + threadIdx.x;
  if (i < n) p[i] = v;
}

// ---------------------------------------------------------------------------
extern "C" void kernel_launch(void* const* d_in, const int* in_sizes, int n_in,
                              void* d_out, int out_size, void* d_ws,
                              size_t ws_size, hipStream_t stream) {
  (void)in_sizes; (void)n_in; (void)out_size; (void)ws_size;
  const int NA = 50000, NC = 100000;
  const int Nn[2] = {NA, NC};
  const int EC[3] = {500000, 500000, 1000000};
  const int esrc[3] = {1, 0, 1};  // 0=answer, 1=concept
  const int edst[3] = {0, 1, 1};

  const float* x_a = (const float*)d_in[0];
  const float* x_c = (const float*)d_in[1];
  const int* ei[3] = {(const int*)d_in[2], (const int*)d_in[3],
                      (const int*)d_in[4]};
  auto P = [&](int i) { return (const float*)d_in[i]; };
  // params (dict insertion order): in_W_a, in_b_a, in_W_c, in_b_c,
  // then per layer: [answer: kW,kb,qW,qb,vW,vb,aW,ab,skip][concept: same]
  // [ei0: arel,mrel,prel][ei1][ei2]
  const int IN_W[2] = {5, 7}, IN_B[2] = {6, 8};
  auto NTp  = [&](int l, int c) { return 9 + l * 27 + c * 9; };
  auto RELp = [&](int l, int e) { return 9 + l * 27 + 18 + e * 3; };

  // ---- workspace carve-up ----
  char* base = (char*)d_ws;
  size_t off = 0;
  auto alloc = [&](size_t bytes) -> char* {
    char* p = base + off;
    off += (bytes + 255) & ~(size_t)255;
    return p;
  };
  float* h[2]  = {(float*)alloc((size_t)NA * 512), (float*)alloc((size_t)NC * 512)};
  float* h2[2] = {(float*)alloc((size_t)NA * 512), (float*)alloc((size_t)NC * 512)};
  float* q[2]  = {(float*)alloc((size_t)NA * 512), (float*)alloc((size_t)NC * 512)};
  float* k[2]  = {(float*)alloc((size_t)NA * 512), (float*)alloc((size_t)NC * 512)};
  float* v[2]  = {(float*)alloc((size_t)NA * 512), (float*)alloc((size_t)NC * 512)};
  float* krel  = (float*)alloc((size_t)NC * 512);
  float* vrel  = (float*)alloc((size_t)NC * 512);
  float* alphaB = (float*)alloc((size_t)EC[2] * 16);
  unsigned* mEnc = (unsigned*)alloc((size_t)NC * 16);
  float* sbuf = (float*)alloc((size_t)NC * 16);
  float* agg[2] = {(float*)alloc((size_t)NA * 512), (float*)alloc((size_t)NC * 512)};
  unsigned short* inWp[2];
  for (int c = 0; c < 2; ++c) inWp[c] = (unsigned short*)alloc(32768 * 2);
  unsigned short* Wp[2][2][4];  // [layer][nt][k,q,v,a]
  for (int l = 0; l < 2; ++l)
    for (int c = 0; c < 2; ++c)
      for (int w = 0; w < 4; ++w) Wp[l][c][w] = (unsigned short*)alloc(16384 * 2);
  unsigned short* relPk[2][3][2];  // [layer][ei][arel/mrel], 4096 bf16 each
  for (int l = 0; l < 2; ++l)
    for (int e = 0; e < 3; ++e)
      for (int m = 0; m < 2; ++m) relPk[l][e][m] = (unsigned short*)alloc(4096 * 2);

  // ---- pack weights to bf16 WMMA fragment order ----
  for (int c = 0; c < 2; ++c)
    pack_w_bf16<<<64, 32, 0, stream>>>(P(IN_W[c]), inWp[c], 128);
  for (int l = 0; l < 2; ++l)
    for (int c = 0; c < 2; ++c)
      for (int w = 0; w < 4; ++w)
        pack_w_bf16<<<32, 32, 0, stream>>>(P(NTp(l, c) + w * 2), Wp[l][c][w], 128);
  for (int l = 0; l < 2; ++l)
    for (int e = 0; e < 3; ++e)
      for (int m = 0; m < 2; ++m)
        for (int hd = 0; hd < 4; ++hd)
          pack_w_bf16<<<2, 32, 0, stream>>>(P(RELp(l, e) + m) + hd * 1024,
                                            relPk[l][e][m] + hd * 1024, 32);

  auto gemm = [&](const float* A, int K, int M, const unsigned short* Bp,
                  const float* bias, float* out, const float* hOld,
                  const float* skipP, int epi) {
    int strips = M / 16;
    hgt_gemm<<<(strips + 7) / 8, 256, 0, stream>>>(A, K, M, Bp, bias, out,
                                                   hOld, skipP, epi);
  };

  // ---- input linears + ReLU ----
  gemm(x_a, 256, NA, inWp[0], P(IN_B[0]), h[0], nullptr, nullptr, 1);
  gemm(x_c, 256, NC, inWp[1], P(IN_B[1]), h[1], nullptr, nullptr, 1);

  // ---- HGT layers ----
  for (int l = 0; l < 2; ++l) {
    const bool last = (l == 1);
    for (int c = 0; c < 2; ++c) {
      const int pb = NTp(l, c);
      // layer 1 only needs q_answer and k/v_concept (only edge type 0 feeds output)
      if (!last || c == 0)
        gemm(h[c], 128, Nn[c], Wp[l][c][1], P(pb + 3), q[c], nullptr, nullptr, 0);
      if (!last || c == 1) {
        gemm(h[c], 128, Nn[c], Wp[l][c][0], P(pb + 1), k[c], nullptr, nullptr, 0);
        gemm(h[c], 128, Nn[c], Wp[l][c][2], P(pb + 5), v[c], nullptr, nullptr, 0);
      }
    }
    fill_u32<<<(NA * 128 + 255) / 256, 256, 0, stream>>>((unsigned*)agg[0], 0u,
                                                         NA * 128);
    if (!last)
      fill_u32<<<(NC * 128 + 255) / 256, 256, 0, stream>>>((unsigned*)agg[1],
                                                           0u, NC * 128);
    const int nE = last ? 1 : 3;
    for (int t = 0; t < nE; ++t) {
      const int st = esrc[t], dt = edst[t], E = EC[t];
      const int* sp = ei[t];
      const int* dp = ei[t] + E;
      hgt_rel<<<Nn[st] / 16, 256, 0, stream>>>(k[st], v[st], relPk[l][t][0],
                                               relPk[l][t][1], krel, vrel);
      fill_u32<<<(Nn[dt] * 4 + 255) / 256, 256, 0, stream>>>(mEnc, 0u,
                                                             Nn[dt] * 4);
      fill_u32<<<(Nn[dt] * 4 + 255) / 256, 256, 0, stream>>>((unsigned*)sbuf,
                                                             0u, Nn[dt] * 4);
      hgt_edge_alpha<<<(E + 7) / 8, 256, 0, stream>>>(
          sp, dp, q[dt], krel, P(RELp(l, t) + 2), alphaB, mEnc, E);
      hgt_edge_expsum<<<(E * 4 + 255) / 256, 256, 0, stream>>>(dp, alphaB,
                                                               mEnc, sbuf, E);
      hgt_edge_scatter<<<(E + 7) / 8, 256, 0, stream>>>(sp, dp, vrel, alphaB,
                                                        sbuf, agg[dt], E);
    }
    for (int c = 0; c < 2; ++c) {
      if (last && c == 1) continue;  // layer-1 concept output is dead
      const int pb = NTp(l, c);
      gelu_inplace<<<(Nn[c] * 128 + 255) / 256, 256, 0, stream>>>(agg[c],
                                                                  Nn[c] * 128);
      float* outp = (last && c == 0) ? (float*)d_out : h2[c];
      gemm(agg[c], 128, Nn[c], Wp[l][c][3], P(pb + 7), outp, h[c], P(pb + 8), 2);
    }
    for (int c = 0; c < 2; ++c) {
      float* tmp = h[c];
      h[c] = h2[c];
      h2[c] = tmp;
    }
  }
}
